// ResidualVQ_47072841564918
// MI455X (gfx1250) — matrix operand order repaired and checked by reading, not compiled, not run on hardware
//
#include <hip/hip_runtime.h>
#include <hip/hip_bf16.h>

#define Q 8
#define C 1024
#define D 512
#define BN (16 * 2048)          // 32768 rows
#define TM 128                  // rows per block (8 waves x 16) -> 4x less L2 traffic
#define NTHREADS 256
#define NWAVES (NTHREADS / 32)
#define NBLOCKS (BN / TM)       // 256
#define NTILE 16                // codes per tile
#define NTILES (C / NTILE)      // 64
#define KC 16                   // k-chunks of 32 (K = 512)

#define AS1 __attribute__((address_space(1)))
#define AS3 __attribute__((address_space(3)))

#if defined(__has_builtin)
#if __has_builtin(__builtin_amdgcn_global_load_async_to_lds_b128)
#define HAVE_ASYNC_LDS 1
#endif
#endif

typedef __attribute__((ext_vector_type(16))) __bf16 v16bf;
typedef __attribute__((ext_vector_type(8)))  float  v8f;
typedef __attribute__((ext_vector_type(4)))  int    v4i;

union V16U {
    v16bf v;
    unsigned short s[16];
    uint4 q4[2];
};

__device__ __forceinline__ unsigned short f2bf(float f) {
    unsigned int u = __float_as_uint(f);
    u += 0x7FFFu + ((u >> 16) & 1u);   // round-to-nearest-even
    return (unsigned short)(u >> 16);
}

// Issue one 16KB tile (16 codes x 512 dims bf16) global -> LDS.
// Async path: 4 x b128 per thread via ASYNCcnt-tracked loads (no VGPR return).
__device__ __forceinline__ void stage_tile(const uint4* __restrict__ src,
                                           uint4* __restrict__ dst, int tid) {
#ifdef HAVE_ASYNC_LDS
    #pragma unroll
    for (int i = 0; i < (NTILE * D / 8) / NTHREADS; ++i) {  // 1024/256 = 4
        int j = tid + i * NTHREADS;
        __builtin_amdgcn_global_load_async_to_lds_b128(
            (AS1 v4i*)(src + j), (AS3 v4i*)(dst + j), 0, 0);
    }
#else
    for (int j = tid; j < NTILE * D / 8; j += NTHREADS) dst[j] = src[j];
#endif
}

__device__ __forceinline__ void wait_async_lds() {
#ifdef HAVE_ASYNC_LDS
#if __has_builtin(__builtin_amdgcn_s_wait_asynccnt)
    __builtin_amdgcn_s_wait_asynccnt(0);
#else
    asm volatile("s_wait_asynccnt 0x0" ::: "memory");
#endif
#endif
}

// ---------------------------------------------------------------------------
// Prep: fp32 codebooks -> bf16 copy in ws, plus halfnorm[c] = 0.5 * ||c||^2
// ---------------------------------------------------------------------------
__global__ __launch_bounds__(64) void rvq_prep(const float* __restrict__ cb,
                                               unsigned short* __restrict__ cb_bf,
                                               float* __restrict__ halfnorm) {
    int g = blockIdx.x;  // code row index 0 .. Q*C-1
    const float* row = cb + (size_t)g * D;
    unsigned short* orow = cb_bf + (size_t)g * D;
    float s = 0.f;
    for (int i = threadIdx.x; i < D; i += 64) {
        float f = row[i];
        orow[i] = f2bf(f);
        s += f * f;
    }
    for (int off = 16; off; off >>= 1) s += __shfl_down(s, off, 32);
    __shared__ float wsum[2];
    if ((threadIdx.x & 31) == 0) wsum[threadIdx.x >> 5] = s;
    __syncthreads();
    if (threadIdx.x == 0) halfnorm[g] = 0.5f * (wsum[0] + wsum[1]);
}

// ---------------------------------------------------------------------------
// Main: fused distance-GEMM (bf16 WMMA) + argmin + gather + residual update
// ---------------------------------------------------------------------------
__global__ __launch_bounds__(NTHREADS) void rvq_main(
        const float* __restrict__ x,
        const float* __restrict__ cb,
        const unsigned short* __restrict__ cb_bf,
        const float* __restrict__ halfnorm,
        float* __restrict__ out_q,
        float* __restrict__ out_idx,
        float* __restrict__ partial) {
    __shared__ float resid[TM][D];                               // 256 KB fp32 residual
    __shared__ __align__(16) unsigned short btile[2][NTILE][D];  // 2 x 16 KB bf16 tile
    __shared__ int rowidx[TM];
    __shared__ float wsum[NWAVES];

    const int tid  = threadIdx.x;
    const int lane = tid & 31;
    const int wv   = tid >> 5;       // wave id 0..7 -> rows [wv*16, wv*16+16)
    const int sub  = lane >> 4;      // half-wave
    const int ln   = lane & 15;
    const size_t base = (size_t)blockIdx.x * TM * D;

    // residual := x   (coalesced float4)
    {
        const float4* xs = (const float4*)(x + base);
        float4* rd = (float4*)&resid[0][0];
        for (int i = tid; i < TM * D / 4; i += NTHREADS) rd[i] = xs[i];
    }
    __syncthreads();

    for (int q = 0; q < Q; ++q) {
        const unsigned short* cbq = cb_bf + (size_t)q * C * D;
        const float* hnq = halfnorm + q * C;

        // Kick off tile 0 staging (async overlaps with A-fragment build).
        stage_tile((const uint4*)cbq, (uint4*)&btile[0][0][0], tid);

        // Build A fragments (16x32 bf16 each) for this wave's 16 rows.
        // A layout (ISA 7.12.2): lane half sub: elems 0..7 -> K = 8*sub+e,
        // elems 8..15 -> K = 16 + 8*sub + (e-8), row M = ln.
        V16U a[KC];
        const float* rrow = &resid[wv * 16 + ln][0];
        #pragma unroll
        for (int kc = 0; kc < KC; ++kc) {
            int b0 = kc * 32 + 8 * sub;
            #pragma unroll
            for (int e = 0; e < 8; ++e) a[kc].s[e] = f2bf(rrow[b0 + e]);
            #pragma unroll
            for (int e = 0; e < 8; ++e) a[kc].s[8 + e] = f2bf(rrow[b0 + 16 + e]);
        }

        float best[8];
        int   bidx[8];
        #pragma unroll
        for (int i = 0; i < 8; ++i) { best[i] = -3.4e38f; bidx[i] = 0; }

        wait_async_lds();
        __syncthreads();

        int buf = 0;
        for (int ct = 0; ct < NTILES; ++ct) {
            // Prefetch next tile into the other buffer while computing.
            if (ct + 1 < NTILES)
                stage_tile((const uint4*)(cbq + (size_t)(ct + 1) * NTILE * D),
                           (uint4*)&btile[buf ^ 1][0][0], tid);

            v8f acc = {0.f, 0.f, 0.f, 0.f, 0.f, 0.f, 0.f, 0.f};
            #pragma unroll
            for (int kc = 0; kc < KC; ++kc) {
                // B layout: lane holds column N = ln, K = 8*sub + {0..7}, 16 + 8*sub + {0..7}.
                V16U bv;
                const uint4* bl = (const uint4*)&btile[buf][ln][kc * 32 + 8 * sub];
                bv.q4[0] = bl[0];
                bv.q4[1] = bl[2];
                acc = __builtin_amdgcn_wmma_f32_16x16x32_bf16(
                    false, a[kc].v, false, bv.v, (short)0, acc, false, false);
            }
            // score = r.c - 0.5*||c||^2 ; argmin distance == argmax score
            float hn = hnq[ct * NTILE + ln];
            int cidx = ct * NTILE + ln;
            #pragma unroll
            for (int i = 0; i < 8; ++i) {  // slot i -> row (8*sub + i)
                float sc = acc[i] - hn;
                if (sc > best[i] || (sc == best[i] && cidx < bidx[i])) {
                    best[i] = sc;
                    bidx[i] = cidx;
                }
            }

            wait_async_lds();
            __syncthreads();
            buf ^= 1;
        }

        // Argmax butterfly over the 16 columns held by lanes within each half.
        #pragma unroll
        for (int off = 1; off < 16; off <<= 1) {
            #pragma unroll
            for (int i = 0; i < 8; ++i) {
                float os = __shfl_xor(best[i], off, 32);
                int   oi = __shfl_xor(bidx[i], off, 32);
                if (os > best[i] || (os == best[i] && oi < bidx[i])) {
                    best[i] = os;
                    bidx[i] = oi;
                }
            }
        }
        if (ln == 0) {
            #pragma unroll
            for (int i = 0; i < 8; ++i) {
                int r = wv * 16 + sub * 8 + i;
                rowidx[r] = bidx[i];
                size_t grow = (size_t)blockIdx.x * TM + r;
                out_idx[grow * Q + q] = (float)bidx[i];
            }
        }
        __syncthreads();

        // Gather winning codewords (fp32), update residual, accumulate loss.
        float lsum = 0.f;
        const float* cbf = cb + (size_t)q * C * D;
        for (int e = tid; e < TM * D; e += NTHREADS) {
            int r = e >> 9, d = e & (D - 1);
            float cv = cbf[(size_t)rowidx[r] * D + d];
            float rv = resid[r][d];
            float df = cv - rv;                 // (q_hard - residual)
            lsum += df * df;
            resid[r][d] = rv - cv;              // new residual
        }
        for (int off = 16; off; off >>= 1) lsum += __shfl_down(lsum, off, 32);
        if (lane == 0) wsum[wv] = lsum;
        __syncthreads();
        if (tid == 0) {
            float t = 0.f;
            #pragma unroll
            for (int w = 0; w < NWAVES; ++w) t += wsum[w];
            partial[q * NBLOCKS + blockIdx.x] = t;
        }
        __syncthreads();
    }

    // quantized_out = x - final residual
    {
        const float4* xs = (const float4*)(x + base);
        const float4* rd = (const float4*)&resid[0][0];
        float4* os = (float4*)(out_q + base);
        for (int i = tid; i < TM * D / 4; i += NTHREADS) {
            float4 xv = xs[i], rv = rd[i];
            os[i] = make_float4(xv.x - rv.x, xv.y - rv.y, xv.z - rv.z, xv.w - rv.w);
        }
    }
}

// ---------------------------------------------------------------------------
// Finalize: deterministic reduction of per-block partials -> vq / qq losses
// ---------------------------------------------------------------------------
__global__ __launch_bounds__(256) void rvq_finalize(const float* __restrict__ partial,
                                                    float* __restrict__ vqs,
                                                    float* __restrict__ qqs) {
    int q = blockIdx.x;
    float s = 0.f;
    for (int i = threadIdx.x; i < NBLOCKS; i += 256) s += partial[q * NBLOCKS + i];
    for (int off = 16; off; off >>= 1) s += __shfl_down(s, off, 32);
    __shared__ float ws[8];
    if ((threadIdx.x & 31) == 0) ws[threadIdx.x >> 5] = s;
    __syncthreads();
    if (threadIdx.x == 0) {
        float t = 0.f;
        for (int i = 0; i < 8; ++i) t += ws[i];
        float mean = t / (float)((size_t)BN * D);
        vqs[q] = mean;
        qqs[q] = 0.25f * mean;
    }
}

// ---------------------------------------------------------------------------
extern "C" void kernel_launch(void* const* d_in, const int* in_sizes, int n_in,
                              void* d_out, int out_size, void* d_ws, size_t ws_size,
                              hipStream_t stream) {
    const float* x  = (const float*)d_in[0];   // [16,2048,512]
    const float* cb = (const float*)d_in[1];   // [8,1024,512]

    // workspace layout
    unsigned short* cb_bf = (unsigned short*)d_ws;                          // 8 MB
    float* halfnorm = (float*)((char*)d_ws + (size_t)Q * C * D * 2);        // 32 KB
    float* partial  = halfnorm + Q * C;                                     // 8 KB

    // output layout: qout | indices | vqs | qqs
    float* out      = (float*)d_out;
    float* out_qout = out;                        // 16,777,216
    float* out_idx  = out + (size_t)BN * D;       // 262,144
    float* out_vqs  = out_idx + (size_t)BN * Q;   // 8
    float* out_qqs  = out_vqs + Q;                // 8

    rvq_prep<<<Q * C, 64, 0, stream>>>(cb, cb_bf, halfnorm);
    rvq_main<<<NBLOCKS, NTHREADS, 0, stream>>>(x, cb, cb_bf, halfnorm,
                                               out_qout, out_idx, partial);
    rvq_finalize<<<Q, 256, 0, stream>>>(partial, out_vqs, out_qqs);
}
